// GATModelv2_5394478924045
// MI455X (gfx1250) — matrix operand order repaired and checked by reading, not compiled, not run on hardware
//
#include <hip/hip_runtime.h>

typedef __attribute__((ext_vector_type(16))) _Float16 v16h;
typedef __attribute__((ext_vector_type(8)))  _Float16 v8h;
typedef __attribute__((ext_vector_type(8)))  float    v8f;

#define NFEAT 64
#define HEADS 8
#define CHAN  8

// ---------- ordered-uint encoding of f32 so atomicMax(u32) == float max ----------
__device__ __forceinline__ unsigned enc_f32(float f) {
  int s = __float_as_int(f);
  return (s >= 0) ? ((unsigned)s + 0x80000000u) : ~(unsigned)s;
}
__device__ __forceinline__ float dec_f32(unsigned k) {
  return (k >= 0x80000000u) ? __int_as_float((int)(k - 0x80000000u))
                            : __int_as_float((int)~k);
}

// =====================================================================
// Elementwise f32 -> f16 convert (for x -> x16, pre-GEMM A operand)
// =====================================================================
__global__ void cvt_f16_kernel(const float* __restrict__ src,
                               _Float16* __restrict__ dst, int n) {
  const int i = blockIdx.x * blockDim.x + threadIdx.x;
  if (i < n) dst[i] = (_Float16)src[i];
}

// =====================================================================
// Weight prep: W [K,64] f32 row-major  ->  Wt [64,K] f16 (transposed)
// Makes GEMM B-fragments contiguous: 16 halves = 2x b128 loads per lane.
// =====================================================================
__global__ void wt_f16_kernel(const float* __restrict__ W,
                              _Float16* __restrict__ Wt, int K) {
  const int i = blockIdx.x * blockDim.x + threadIdx.x;
  if (i >= K * 64) return;
  const int kk = i >> 6;
  const int c  = i & 63;
  Wt[(size_t)c * K + kk] = (_Float16)W[i];
}

// =====================================================================
// WMMA GEMM: C0 = A @ B0 [, C1 = A @ B1] (DUAL compile-time).
// A16:[N,K] f16 row-major; Bt:[64,K] f16 transposed; C:[N,64] f32.
// 256 threads = 8 waves; wave w owns rows blockIdx*128 + w*16, ALL 64 cols
// (4 col-tiles): A fragment loaded once per k-step, feeds 4 (8 dual) WMMAs.
// All operand fetches are global_load_b128; no per-element converts.
// =====================================================================
template <bool DUAL>
__global__ void gemm_wmma_kernel(const _Float16* __restrict__ A16,
                                 const _Float16* __restrict__ Bt0,
                                 const _Float16* __restrict__ Bt1,
                                 float* __restrict__ C0,
                                 float* __restrict__ C1,
                                 int N, int K) {
  const int lane = threadIdx.x & 31;
  const int wave = threadIdx.x >> 5;
  const int row0 = blockIdx.x * 128 + wave * 16;
  if (row0 >= N) return;                 // N % 16 == 0: whole-tile guard
  const bool hi  = lane >= 16;
  const int r    = lane & 15;

  v8f acc0[4] = {v8f{}, v8f{}, v8f{}, v8f{}};
  v8f acc1[4] = {v8f{}, v8f{}, v8f{}, v8f{}};

  for (int k = 0; k < K; k += 32) {
    // ---- A fragment (ISA 7.12.2, 16-bit A 16x32):
    //   lane<16 -> K {k..k+7, k+16..k+23} of row row0+r
    //   lane>=16 -> K {k+8..k+15, k+24..k+31}
    const _Float16* ap = A16 + (size_t)(row0 + r) * K + k + (hi ? 8 : 0);
    v8h alo = *(const v8h*)ap;
    v8h ahi = *(const v8h*)(ap + 16);
    v16h a = __builtin_shufflevector(alo, ahi, 0,1,2,3,4,5,6,7,
                                               8,9,10,11,12,13,14,15);
    const int kb = k + (hi ? 8 : 0);
    // ---- hoist all B loads for this k-step so WMMAs can pipeline ----
    v16h b0[4], b1[4];
#pragma unroll
    for (int ct = 0; ct < 4; ++ct) {
      const int col = ct * 16 + r;       // lane holds column `col`, K rows kb..kb+7, kb+16..kb+23
      const _Float16* bp = Bt0 + (size_t)col * K + kb;
      b0[ct] = __builtin_shufflevector(*(const v8h*)bp, *(const v8h*)(bp + 16),
                                       0,1,2,3,4,5,6,7,8,9,10,11,12,13,14,15);
      if (DUAL) {
        const _Float16* bq = Bt1 + (size_t)col * K + kb;
        b1[ct] = __builtin_shufflevector(*(const v8h*)bq, *(const v8h*)(bq + 16),
                                         0,1,2,3,4,5,6,7,8,9,10,11,12,13,14,15);
      }
    }
#pragma unroll
    for (int ct = 0; ct < 4; ++ct) {
      acc0[ct] = __builtin_amdgcn_wmma_f32_16x16x32_f16(
          false, a, false, b0[ct], (short)0, acc0[ct], false, false);
      if (DUAL)
        acc1[ct] = __builtin_amdgcn_wmma_f32_16x16x32_f16(
            false, a, false, b1[ct], (short)0, acc1[ct], false, false);
    }
  }
  // ---- C store: VGPR rr holds row rr (lanes 0-15) / row rr+8 (lanes 16-31)
#pragma unroll
  for (int ct = 0; ct < 4; ++ct) {
#pragma unroll
    for (int rr = 0; rr < 8; ++rr) {
      const int row = row0 + rr + (hi ? 8 : 0);
      C0[(size_t)row * NFEAT + ct * 16 + r] = acc0[ct][rr];
      if (DUAL) C1[(size_t)row * NFEAT + ct * 16 + r] = acc1[ct][rr];
    }
  }
}

// =====================================================================
// Edge pass 1: score[e,h] = att[h,:]·leakyrelu(xl[src]+xr[dst]); seg-max(dst)
// =====================================================================
__global__ void edge_score_kernel(const float* __restrict__ xl,
                                  const float* __restrict__ xr,
                                  const long long* __restrict__ ei,
                                  const float* __restrict__ att,
                                  float* __restrict__ p,
                                  unsigned* __restrict__ smax,
                                  int Eorig, int Etot) {
  const int e = blockIdx.x * blockDim.x + threadIdx.x;
  if (e >= Etot) return;
  int s, d;
  if (e < Eorig) { s = (int)ei[e]; d = (int)ei[(size_t)Eorig + e]; }
  else           { s = d = e - Eorig; }
  const float* lrow = xl + (size_t)s * NFEAT;
  const float* rrow = xr + (size_t)d * NFEAT;
#pragma unroll
  for (int hh = 0; hh < HEADS; ++hh) {
    float4 l0 = *(const float4*)(lrow + hh * CHAN);
    float4 l1 = *(const float4*)(lrow + hh * CHAN + 4);
    float4 r0 = *(const float4*)(rrow + hh * CHAN);
    float4 r1 = *(const float4*)(rrow + hh * CHAN + 4);
    float lv[8] = {l0.x + r0.x, l0.y + r0.y, l0.z + r0.z, l0.w + r0.w,
                   l1.x + r1.x, l1.y + r1.y, l1.z + r1.z, l1.w + r1.w};
    float sc = 0.f;
#pragma unroll
    for (int i = 0; i < 8; ++i) {
      float v = lv[i];
      v = v > 0.f ? v : 0.2f * v;               // leaky_relu
      sc += att[hh * CHAN + i] * v;
    }
    p[(size_t)e * HEADS + hh] = sc;
    atomicMax(&smax[(size_t)d * HEADS + hh], enc_f32(sc));
  }
}

// =====================================================================
// Edge pass 2: p = exp(score - smax[dst]); denom[dst] += p
// =====================================================================
__global__ void edge_expsum_kernel(const long long* __restrict__ ei,
                                   float* __restrict__ p,
                                   const unsigned* __restrict__ smax,
                                   float* __restrict__ denom,
                                   int Eorig, int Etot) {
  const int e = blockIdx.x * blockDim.x + threadIdx.x;
  if (e >= Etot) return;
  const int d = (e < Eorig) ? (int)ei[(size_t)Eorig + e] : e - Eorig;
#pragma unroll
  for (int hh = 0; hh < HEADS; ++hh) {
    const float m  = dec_f32(smax[(size_t)d * HEADS + hh]);
    const float pe = __expf(p[(size_t)e * HEADS + hh] - m);
    p[(size_t)e * HEADS + hh] = pe;
    atomicAdd(&denom[(size_t)d * HEADS + hh], pe);
  }
}

// =====================================================================
// Edge pass 3: g[dst] += xl[src] * (p / (denom[dst] + 1e-16))
// =====================================================================
__global__ void edge_aggregate_kernel(const float* __restrict__ xl,
                                      const long long* __restrict__ ei,
                                      const float* __restrict__ p,
                                      const float* __restrict__ denom,
                                      float* __restrict__ g,
                                      int Eorig, int Etot) {
  const int e = blockIdx.x * blockDim.x + threadIdx.x;
  if (e >= Etot) return;
  int s, d;
  if (e < Eorig) { s = (int)ei[e]; d = (int)ei[(size_t)Eorig + e]; }
  else           { s = d = e - Eorig; }
  const float* lrow = xl + (size_t)s * NFEAT;
  float* grow = g + (size_t)d * NFEAT;
#pragma unroll
  for (int hh = 0; hh < HEADS; ++hh) {
    const float alpha = p[(size_t)e * HEADS + hh] /
                        (denom[(size_t)d * HEADS + hh] + 1e-16f);
    float4 l0 = *(const float4*)(lrow + hh * CHAN);
    float4 l1 = *(const float4*)(lrow + hh * CHAN + 4);
    float lv[8] = {l0.x, l0.y, l0.z, l0.w, l1.x, l1.y, l1.z, l1.w};
#pragma unroll
    for (int i = 0; i < 8; ++i)
      atomicAdd(grow + hh * CHAN + i, lv[i] * alpha);
  }
}

// =====================================================================
// BN stats: accum[ch] = sum, accum[64+ch] = sumsq  (64 channels)
// =====================================================================
__global__ void bn_stats_kernel(const float* __restrict__ g,
                                float* __restrict__ accum, int N) {
  __shared__ float ls[256], lss[256];
  const int ch  = threadIdx.x & 63;
  const int grp = threadIdx.x >> 6;  // 0..3
  float s = 0.f, ss = 0.f;
  for (int row = blockIdx.x * 4 + grp; row < N; row += gridDim.x * 4) {
    const float v = g[(size_t)row * NFEAT + ch];
    s += v; ss += v * v;
  }
  ls[threadIdx.x] = s; lss[threadIdx.x] = ss;
  __syncthreads();
  if (grp == 0) {
#pragma unroll
    for (int j = 1; j < 4; ++j) { s += ls[j * 64 + ch]; ss += lss[j * 64 + ch]; }
    atomicAdd(&accum[ch], s);
    atomicAdd(&accum[64 + ch], ss);
  }
}

// =====================================================================
// BN apply + optional residual + ELU; writes f32 h AND f16 mirror h16
// (h16 feeds the next layer's WMMA GEMM with b128 loads, no converts)
// =====================================================================
__global__ void bn_apply_kernel(const float* __restrict__ g,
                                const float* __restrict__ accum,
                                const float* __restrict__ gamma,
                                const float* __restrict__ beta,
                                float* __restrict__ h,
                                _Float16* __restrict__ h16,
                                int N, int with_residual) {
  const int i = blockIdx.x * blockDim.x + threadIdx.x;
  if (i >= N * NFEAT) return;
  const int ch = i & 63;
  const float invN = 1.0f / (float)N;
  const float mu  = accum[ch] * invN;
  const float var = fmaxf(accum[64 + ch] * invN - mu * mu, 0.f);
  float y = (g[i] - mu) * rsqrtf(var + 1e-5f) * gamma[ch] + beta[ch];
  if (with_residual) y += h[i];
  y = y > 0.f ? y : __expf(y) - 1.f;  // elu
  h[i]   = y;
  h16[i] = (_Float16)y;
}

// =====================================================================
// Final projection: out[n,:] = h[n,:] @ post_w + post_b  (64 -> 2)
// =====================================================================
__global__ void post_kernel(const float* __restrict__ h,
                            const float* __restrict__ w,
                            const float* __restrict__ b,
                            float* __restrict__ out, int N) {
  const int n = blockIdx.x * blockDim.x + threadIdx.x;
  if (n >= N) return;
  float a0 = b[0], a1 = b[1];
  const float* hr = h + (size_t)n * NFEAT;
#pragma unroll
  for (int c = 0; c < NFEAT; ++c) {
    const float v = hr[c];
    a0 += v * w[c * 2 + 0];
    a1 += v * w[c * 2 + 1];
  }
  out[(size_t)n * 2 + 0] = a0;
  out[(size_t)n * 2 + 1] = a1;
}

extern "C" void kernel_launch(void* const* d_in, const int* in_sizes, int n_in,
                              void* d_out, int out_size, void* d_ws, size_t ws_size,
                              hipStream_t stream) {
  const float*     x      = (const float*)d_in[0];
  const long long* ei     = (const long long*)d_in[1];   // int64 [2,E]
  const float*     pre_w  = (const float*)d_in[2];
  const float*     lin_l  = (const float*)d_in[3];
  const float*     lin_r  = (const float*)d_in[4];
  const float*     att_w  = (const float*)d_in[5];
  const float*     gamma  = (const float*)d_in[6];
  const float*     beta   = (const float*)d_in[7];
  const float*     post_w = (const float*)d_in[8];
  const float*     post_b = (const float*)d_in[9];

  const int N     = in_sizes[0] / 256;   // 100000
  const int Eorig = in_sizes[1] / 2;     // 1600000
  const int Etot  = Eorig + N;

  // ---- workspace carve-out (~176 MB; all offsets 32B multiples) ----
  float* ws = (float*)d_ws;
  float* h     = ws; ws += (size_t)N * NFEAT;
  float* xl    = ws; ws += (size_t)N * NFEAT;
  float* xr    = ws; ws += (size_t)N * NFEAT;
  float* g     = ws; ws += (size_t)N * NFEAT;
  float* denom = ws; ws += (size_t)N * HEADS;
  unsigned* smax = (unsigned*)ws; ws += (size_t)N * HEADS;
  float* p     = ws; ws += (size_t)Etot * HEADS;
  float* accum = ws; ws += 128;
  _Float16* hwp    = (_Float16*)ws;
  _Float16* h16    = hwp; hwp += (size_t)N * NFEAT;     // f16 mirror of h
  _Float16* pre_wt = hwp; hwp += 64 * 256;
  _Float16* lwl    = hwp; hwp += 4 * 64 * 64;
  _Float16* lwr    = hwp; hwp += 4 * 64 * 64;
  // x16 (N*256 halves = 51.2 MB) aliases the p buffer (54.4 MB), which is
  // dead until the first edge pass — pre-GEMM finishes before p is written.
  _Float16* x16 = (_Float16*)p;

  const dim3 blk(256);
  const int gemm_blocks = (N + 127) / 128;
  const int elem_blocks = (N * NFEAT + 255) / 256;
  const int edge_blocks = (Etot + 255) / 256;
  const int node_blocks = (N + 255) / 256;

  // ---- operand prep (f16 mirrors / transposed weights) ----
  cvt_f16_kernel<<<(N * 256 + 255) / 256, blk, 0, stream>>>(x, x16, N * 256);
  wt_f16_kernel<<<(256 * 64 + 255) / 256, blk, 0, stream>>>(pre_w, pre_wt, 256);
  for (int l = 0; l < 4; ++l) {
    wt_f16_kernel<<<(64 * 64 + 255) / 256, blk, 0, stream>>>(
        lin_l + (size_t)l * NFEAT * NFEAT, lwl + (size_t)l * NFEAT * NFEAT, 64);
    wt_f16_kernel<<<(64 * 64 + 255) / 256, blk, 0, stream>>>(
        lin_r + (size_t)l * NFEAT * NFEAT, lwr + (size_t)l * NFEAT * NFEAT, 64);
  }

  // stage 0: h = elu(bn(x @ pre_w))
  gemm_wmma_kernel<false><<<gemm_blocks, blk, 0, stream>>>(
      x16, pre_wt, nullptr, g, nullptr, N, 256);
  hipMemsetAsync(accum, 0, 128 * sizeof(float), stream);
  bn_stats_kernel<<<512, blk, 0, stream>>>(g, accum, N);
  bn_apply_kernel<<<elem_blocks, blk, 0, stream>>>(g, accum, gamma, beta, h, h16, N, 0);

  for (int layer = 0; layer < 4; ++layer) {
    const _Float16* Wl = lwl + (size_t)layer * NFEAT * NFEAT;
    const _Float16* Wr = lwr + (size_t)layer * NFEAT * NFEAT;
    const float*   att = att_w + (size_t)layer * HEADS * CHAN;

    // xl = h@Wl, xr = h@Wr in one pass (h16 read once; both fit in L2)
    gemm_wmma_kernel<true><<<gemm_blocks, blk, 0, stream>>>(
        h16, Wl, Wr, xl, xr, N, NFEAT);

    hipMemsetAsync(smax,  0, (size_t)N * HEADS * sizeof(unsigned), stream);
    hipMemsetAsync(denom, 0, (size_t)N * HEADS * sizeof(float), stream);
    hipMemsetAsync(g,     0, (size_t)N * NFEAT * sizeof(float), stream);

    edge_score_kernel    <<<edge_blocks, blk, 0, stream>>>(xl, xr, ei, att, p, smax, Eorig, Etot);
    edge_expsum_kernel   <<<edge_blocks, blk, 0, stream>>>(ei, p, smax, denom, Eorig, Etot);
    edge_aggregate_kernel<<<edge_blocks, blk, 0, stream>>>(xl, ei, p, denom, g, Eorig, Etot);

    hipMemsetAsync(accum, 0, 128 * sizeof(float), stream);
    bn_stats_kernel<<<512, blk, 0, stream>>>(g, accum, N);
    bn_apply_kernel<<<elem_blocks, blk, 0, stream>>>(
        g, accum, gamma + (size_t)(layer + 1) * NFEAT,
        beta + (size_t)(layer + 1) * NFEAT, h, h16, N, 1);
  }

  post_kernel<<<node_blocks, blk, 0, stream>>>(h, post_w, post_b, (float*)d_out, N);
}
